// MS_MSA_3693671875050
// MI455X (gfx1250) — matrix operand order repaired and checked by reading, not compiled, not run on hardware
//
#include <hip/hip_runtime.h>
#include <hip/hip_bf16.h>
#include <math.h>

// ---------------------------------------------------------------------------
// MS-MSA on MI455X (gfx1250, wave32, WMMA).
// All dense 256x256 projections run on v_wmma_f32_16x16x32_bf16.
// Gram matrix (channel attention scores) uses split-K WMMA with LDS transpose.
// Depthwise convs / softmax / gating are memory-bound VALU kernels (L2-served).
// ---------------------------------------------------------------------------

typedef __attribute__((ext_vector_type(16))) __bf16 v16bf;
typedef __attribute__((ext_vector_type(8)))  float  v8f;

#define NPIX   131072          // B * H * W
#define HW     65536           // H * W
#define CCH    256

__device__ __forceinline__ unsigned short f2bf(float f) {
    unsigned u = __float_as_uint(f);
    u += 0x7fffu + ((u >> 16) & 1u);           // round-to-nearest-even
    return (unsigned short)(u >> 16);
}
__device__ __forceinline__ float bf2f(unsigned short h) {
    return __uint_as_float(((unsigned)h) << 16);
}

// ---------------------------------------------------------------------------
// Weight prep: f32 [256,256] -> bf16, K-pair-major layout:
//   u16 index = (k>>1)*512 + n*2 + (k&1)   (so (k,k+1) at col n share a dword)
// mats: 0=wq 1=wk 2=wv 3=mm_c1(T) 4=mm_c2(T) 5=proj
// ---------------------------------------------------------------------------
__global__ void wprep_kernel(const float* __restrict__ wq, const float* __restrict__ wk,
                             const float* __restrict__ wv, const float* __restrict__ c1,
                             const float* __restrict__ c2, const float* __restrict__ pj,
                             unsigned short* __restrict__ out)
{
    int idx = blockIdx.x * 256 + threadIdx.x;       // < 6*65536
    int mat = idx >> 16;
    int e   = idx & 65535;
    int kk  = e >> 8, n = e & 255;
    const float* srcs[6] = { wq, wk, wv, c1, c2, pj };
    bool tr = (mat == 3 || mat == 4);               // x @ W^T for the 1x1 convs
    float v = tr ? srcs[mat][n * 256 + kk] : srcs[mat][kk * 256 + n];
    out[(size_t)mat * 65536 + (size_t)(kk >> 1) * 512 + n * 2 + (kk & 1)] = f2bf(v);
}

// ---------------------------------------------------------------------------
// Tiled WMMA GEMM:  O[M,256] = A[M,256] * B(prepped bf16) (+bias)
// BM=128 rows/block, BN cols/block, 8 waves as 4(M) x 2(N), wave tile 32 x BN/2.
// B slice lives in LDS for the whole block; A tiles double-buffered.
// ---------------------------------------------------------------------------
template <int NMAT, bool A_F32, bool OUT_F32, bool BIAS, int BN>
__global__ __launch_bounds__(256)
void gemm_wmma(const void* __restrict__ Av,
               const unsigned short* __restrict__ B0,
               const unsigned short* __restrict__ B1,
               const unsigned short* __restrict__ B2,
               void* __restrict__ O0, void* __restrict__ O1, void* __restrict__ O2,
               const float* __restrict__ bias0, const float* __restrict__ bias1,
               const float* __restrict__ bias2)
{
    constexpr int KSTEPS = 8;            // K=256 in chunks of 32
    constexpr int NTJ    = BN / 32;      // 16-col B tiles per wave
    extern __shared__ __align__(16) unsigned int sm[];
    unsigned int* LB = sm;                               // NMAT*128*BN dwords
    unsigned int* LA = sm + NMAT * 128 * BN;             // 2 * 2048 dwords

    const int t    = threadIdx.x;
    const int lane = t & 31, wave = t >> 5;
    const int half = lane >> 4, r = lane & 15;
    const int gm   = blockIdx.x * 128;
    const int n0   = blockIdx.y * BN;

    const unsigned short* Bs[3] = { B0, B1, B2 };
    void*        Os[3]    = { O0, O1, O2 };
    const float* biasv[3] = { bias0, bias1, bias2 };

    // ---- stage the full K x BN slice of each B matrix (pair-major) ----
    for (int i = t; i < NMAT * 128 * (BN / 4); i += 256) {
        int mat  = i / (128 * (BN / 4));
        int rem  = i - mat * (128 * (BN / 4));
        int pair = rem / (BN / 4);
        int c4   = rem - pair * (BN / 4);
        const uint4* src = (const uint4*)(Bs[mat] + (size_t)pair * 512 + (size_t)(n0 + c4 * 4) * 2);
        *((uint4*)(LB + (size_t)mat * 128 * BN + pair * BN + c4 * 4)) = *src;
    }

    auto stageA = [&](int kk, int buf) {
        int row = t >> 1;
        int cb  = (t & 1) * 16;                       // element col base within 32
        unsigned int* dst = LA + buf * 2048 + row * 16 + (cb >> 1);
        if constexpr (A_F32) {
            const float* srow = (const float*)Av + (size_t)(gm + row) * 256 + kk * 32 + cb;
            if (kk + 1 < KSTEPS) __builtin_prefetch(srow + 32, 0, 1);
            const float4* src = (const float4*)srow;
            unsigned int o[8];
#pragma unroll
            for (int i = 0; i < 4; ++i) {
                float4 f = src[i];
                o[i*2+0] = (unsigned)f2bf(f.x) | ((unsigned)f2bf(f.y) << 16);
                o[i*2+1] = (unsigned)f2bf(f.z) | ((unsigned)f2bf(f.w) << 16);
            }
            uint4 u0; u0.x=o[0]; u0.y=o[1]; u0.z=o[2]; u0.w=o[3];
            uint4 u1; u1.x=o[4]; u1.y=o[5]; u1.z=o[6]; u1.w=o[7];
            ((uint4*)dst)[0] = u0; ((uint4*)dst)[1] = u1;
        } else {
            const unsigned short* srow = (const unsigned short*)Av + (size_t)(gm + row) * 256 + kk * 32 + cb;
            if (kk + 1 < KSTEPS) __builtin_prefetch(srow + 32, 0, 1);
            const uint4* src = (const uint4*)srow;
            ((uint4*)dst)[0] = src[0]; ((uint4*)dst)[1] = src[1];
        }
    };

    v8f acc[NMAT][2][NTJ];
#pragma unroll
    for (int m = 0; m < NMAT; ++m)
#pragma unroll
        for (int a = 0; a < 2; ++a)
#pragma unroll
            for (int j = 0; j < NTJ; ++j)
#pragma unroll
                for (int q = 0; q < 8; ++q) acc[m][a][j][q] = 0.0f;

    stageA(0, 0);
    __syncthreads();

    const int mbase = (wave >> 1) * 32;
    const int nbl   = (wave & 1) * (BN / 2);

    for (int kk = 0; kk < KSTEPS; ++kk) {
        if (kk + 1 < KSTEPS) stageA(kk + 1, (kk + 1) & 1);

        const unsigned int* la = LA + (kk & 1) * 2048;
        union { unsigned int u[8]; v16bf v; } af[2];
#pragma unroll
        for (int ti = 0; ti < 2; ++ti) {
            int row = mbase + ti * 16 + r;
#pragma unroll
            for (int j = 0; j < 8; ++j) {
                int pair = (j & 3) + (half << 2) + ((j >> 2) << 3);   // A 16-bit layout
                af[ti].u[j] = la[row * 16 + pair];
            }
        }
#pragma unroll
        for (int mat = 0; mat < NMAT; ++mat) {
#pragma unroll
            for (int tj = 0; tj < NTJ; ++tj) {
                union { unsigned int u[8]; v16bf v; } bfm;
                int col = nbl + tj * 16 + r;
#pragma unroll
                for (int j = 0; j < 8; ++j) {
                    int pair = kk * 16 + (j & 3) + (half << 2) + ((j >> 2) << 3);             // B 16-bit layout
                    bfm.u[j] = LB[(size_t)mat * 128 * BN + pair * BN + col];
                }
#pragma unroll
                for (int ti = 0; ti < 2; ++ti)
                    acc[mat][ti][tj] = __builtin_amdgcn_wmma_f32_16x16x32_bf16(
                        false, af[ti].v, false, bfm.v, (short)0, acc[mat][ti][tj], false, false);
            }
        }
        __syncthreads();
    }

    // ---- write out: C/D layout  m = ti*16 + rr + 8*half, n = lane&15 ----
#pragma unroll
    for (int mat = 0; mat < NMAT; ++mat) {
#pragma unroll
        for (int tj = 0; tj < NTJ; ++tj) {
            int col = n0 + nbl + tj * 16 + r;
            float bv = BIAS ? biasv[mat][col] : 0.0f;
#pragma unroll
            for (int ti = 0; ti < 2; ++ti) {
#pragma unroll
                for (int rr = 0; rr < 8; ++rr) {
                    int row = gm + mbase + ti * 16 + rr + 8 * half;
                    size_t idx = (size_t)row * 256 + col;
                    float v = acc[mat][ti][tj][rr] + bv;
                    if constexpr (OUT_F32) ((float*)Os[mat])[idx] = v;
                    else ((unsigned short*)Os[mat])[idx] = f2bf(v);
                }
            }
        }
    }
}

// ---------------------------------------------------------------------------
// Gram kernel: G[b,h,d,e] = sum_n k[b,n,h*32+d] * q[b,n,h*32+e]
// grid = (b,h) x 64 split-K blocks; each block covers 1024 n.
// Stage [32 ch][256 n] transposed tiles in LDS (pairs along n), 8 waves each
// WMMA a 32-n slice, then deterministic LDS reduce -> Gpart.
// ---------------------------------------------------------------------------
__global__ __launch_bounds__(256)
void gram_wmma(const unsigned short* __restrict__ q,
               const unsigned short* __restrict__ k,
               float* __restrict__ Gpart)
{
    __shared__ unsigned int sm[8192];                    // 32 KB (staging & reduce)
    const int t = threadIdx.x, lane = t & 31, wave = t >> 5;
    const int half = lane >> 4, r = lane & 15;
    const int bh = blockIdx.x >> 6;                      // 0..15
    const int split = blockIdx.x & 63;
    const int b = bh >> 3, h = bh & 7;
    const size_t nstart = (size_t)b * HW + (size_t)split * 1024;

    v8f acc[2][2];
#pragma unroll
    for (int a = 0; a < 2; ++a)
#pragma unroll
        for (int c = 0; c < 2; ++c)
#pragma unroll
            for (int i = 0; i < 8; ++i) acc[a][c][i] = 0.0f;

    for (int c4 = 0; c4 < 4; ++c4) {
        __syncthreads();
        {   // transpose-stage: threads 0..127 -> K, 128..255 -> Q; 2 n-rows each
            int matsel = t >> 7;
            int nl2    = t & 127;
            const unsigned short* src = (matsel ? q : k)
                + (nstart + (size_t)c4 * 256 + (size_t)nl2 * 2) * 256 + h * 32;
            const uint4* s0 = (const uint4*)src;
            const uint4* s1 = (const uint4*)(src + 256);
            uint4 R0[4] = { s0[0], s0[1], s0[2], s0[3] };
            uint4 R1[4] = { s1[0], s1[1], s1[2], s1[3] };
            const unsigned int* a0 = (const unsigned int*)R0;
            const unsigned int* a1 = (const unsigned int*)R1;
            unsigned int* dst = sm + matsel * 4096;      // [32 ch][128 n-pairs]
#pragma unroll
            for (int j = 0; j < 16; ++j) {
                unsigned lo = a0[j], hi = a1[j];
                dst[(2 * j + 0) * 128 + nl2] = (lo & 0xffffu) | (hi << 16);
                dst[(2 * j + 1) * 128 + nl2] = (lo >> 16) | (hi & 0xffff0000u);
            }
        }
        __syncthreads();

        const int pw = wave * 16;                        // this wave's n-pair window
        union { unsigned int u[8]; v16bf v; } af[2], bfm[2];
#pragma unroll
        for (int ti = 0; ti < 2; ++ti) {                 // A = K rows (d)
            int d = ti * 16 + r;
#pragma unroll
            for (int j = 0; j < 8; ++j) {
                int pa = (j & 3) + (half << 2) + ((j >> 2) << 3);
                af[ti].u[j] = sm[d * 128 + pw + pa];
            }
        }
#pragma unroll
        for (int tj = 0; tj < 2; ++tj) {                 // B = Q cols (e)
            int e = tj * 16 + r;
#pragma unroll
            for (int j = 0; j < 8; ++j) {
                int pb = (j & 3) + (half << 2) + ((j >> 2) << 3);
                bfm[tj].u[j] = sm[4096 + e * 128 + pw + pb];
            }
        }
#pragma unroll
        for (int ti = 0; ti < 2; ++ti)
#pragma unroll
            for (int tj = 0; tj < 2; ++tj)
                acc[ti][tj] = __builtin_amdgcn_wmma_f32_16x16x32_bf16(
                    false, af[ti].v, false, bfm[tj].v, (short)0, acc[ti][tj], false, false);
    }

    __syncthreads();
    float* red = (float*)sm;                             // 8 waves x 1024
#pragma unroll
    for (int ti = 0; ti < 2; ++ti)
#pragma unroll
        for (int tj = 0; tj < 2; ++tj)
#pragma unroll
            for (int rr = 0; rr < 8; ++rr) {
                int d = ti * 16 + rr + 8 * half;
                int e = tj * 16 + r;
                red[wave * 1024 + d * 32 + e] = acc[ti][tj][rr];
            }
    __syncthreads();
    for (int i = t; i < 1024; i += 256) {
        float s = 0.0f;
#pragma unroll
        for (int w = 0; w < 8; ++w) s += red[w * 1024 + i];
        Gpart[((size_t)bh * 64 + split) * 1024 + i] = s;
    }
}

// ---------------------------------------------------------------------------
// Row-norm partials: sum over a 1024-n slab of k^2 and q^2 for all 256 channels
// ---------------------------------------------------------------------------
__global__ __launch_bounds__(256)
void norms_kernel(const unsigned short* __restrict__ q,
                  const unsigned short* __restrict__ k,
                  float* __restrict__ Npart)
{
    int b = blockIdx.x >> 6, s = blockIdx.x & 63, c = threadIdx.x;
    size_t base = ((size_t)b * HW + (size_t)s * 1024) * 256 + c;
    float k2 = 0.0f, q2 = 0.0f;
    for (int n = 0; n < 1024; ++n) {
        float kv = bf2f(k[base + (size_t)n * 256]);
        float qv = bf2f(q[base + (size_t)n * 256]);
        k2 = fmaf(kv, kv, k2);
        q2 = fmaf(qv, qv, q2);
    }
    Npart[(((size_t)b * 64 + s) * 2 + 0) * 256 + c] = k2;
    Npart[(((size_t)b * 64 + s) * 2 + 1) * 256 + c] = q2;
}

// ---------------------------------------------------------------------------
// Softmax over e with l2-norm correction and rescale; one (b,h) per block.
// ---------------------------------------------------------------------------
__global__ void softmax_kernel(const float* __restrict__ Gpart,
                               const float* __restrict__ Npart,
                               const float* __restrict__ rescale,
                               float* __restrict__ attn)
{
    __shared__ float nkS[32], nqS[32];
    int bh = blockIdx.x, b = bh >> 3, h = bh & 7;
    int d = threadIdx.x;                                  // 32 threads
    int c = h * 32 + d;
    float nk = 0.0f, nq = 0.0f;
    for (int s = 0; s < 64; ++s) {
        nk += Npart[(((size_t)b * 64 + s) * 2 + 0) * 256 + c];
        nq += Npart[(((size_t)b * 64 + s) * 2 + 1) * 256 + c];
    }
    nkS[d] = fmaxf(sqrtf(nk), 1e-12f);
    nqS[d] = fmaxf(sqrtf(nq), 1e-12f);
    __syncthreads();
    float resc = rescale[h];
    float g[32];
    float invd = resc / nkS[d];
    for (int e = 0; e < 32; ++e) {
        float s_ = 0.0f;
        for (int s = 0; s < 64; ++s)
            s_ += Gpart[((size_t)bh * 64 + s) * 1024 + d * 32 + e];
        g[e] = s_ * invd / nqS[e];
    }
    float mx = -1e30f;
    for (int e = 0; e < 32; ++e) mx = fmaxf(mx, g[e]);
    float sum = 0.0f;
    for (int e = 0; e < 32; ++e) { g[e] = expf(g[e] - mx); sum += g[e]; }
    float inv = 1.0f / sum;
    for (int e = 0; e < 32; ++e) attn[(size_t)bh * 1024 + d * 32 + e] = g[e] * inv;
}

// ---------------------------------------------------------------------------
// MaskGuidedMechanism tail: v *= (m1 * sigmoid(dw5x5(m2)+b) + m1). In-place on v.
// One pixel per block, channel-last (fully coalesced; m2 reuse served by L2).
// ---------------------------------------------------------------------------
__global__ __launch_bounds__(256)
void maskcomb_kernel(const unsigned short* __restrict__ m1,
                     const unsigned short* __restrict__ m2,
                     unsigned short* __restrict__ v,
                     const float* __restrict__ wdw,
                     const float* __restrict__ bdw)
{
    int p = blockIdx.x;
    int b = p >> 16, yx = p & 65535, y = yx >> 8, x = yx & 255;
    int c = threadIdx.x;
    float acc = 0.0f;
#pragma unroll
    for (int dy = -2; dy <= 2; ++dy) {
        int yy = y + dy; if (yy < 0 || yy > 255) continue;
#pragma unroll
        for (int dx = -2; dx <= 2; ++dx) {
            int xx = x + dx; if (xx < 0 || xx > 255) continue;
            acc = fmaf(bf2f(m2[(((size_t)b * HW) + yy * 256 + xx) * 256 + c]),
                       wdw[c * 25 + (dy + 2) * 5 + (dx + 2)], acc);
        }
    }
    float a  = 1.0f / (1.0f + expf(-(acc + bdw[c])));
    size_t idx = (size_t)p * 256 + c;
    float mv = bf2f(m1[idx]);
    float mf = fmaf(mv, a, mv);
    v[idx] = f2bf(bf2f(v[idx]) * mf);
}

// ---------------------------------------------------------------------------
// out[p, h*32+d] = sum_e attn[b,h,d,e] * v[p, h*32+e]   (per-pixel 32x32 matvec)
// attn for this batch staged in LDS; v row loads are wave-broadcast.
// ---------------------------------------------------------------------------
__global__ __launch_bounds__(256)
void attnapply_kernel(const float* __restrict__ attn,
                      const unsigned short* __restrict__ v,
                      unsigned short* __restrict__ out)
{
    __shared__ float aS[8192];
    int p0 = blockIdx.x * 32;
    int b  = p0 >> 16;
    for (int i = threadIdx.x; i < 8192; i += 256) aS[i] = attn[(size_t)b * 8192 + i];
    __syncthreads();
    int h = threadIdx.x >> 5, d = threadIdx.x & 31;
    const float* arow = aS + h * 1024 + d * 32;
    for (int pp = 0; pp < 32; ++pp) {
        size_t pix = (size_t)p0 + pp;
        const uint4* vr = (const uint4*)(v + pix * 256 + h * 32);
        float acc = 0.0f;
#pragma unroll
        for (int i = 0; i < 4; ++i) {
            uint4 u = vr[i];
            unsigned int w[4] = { u.x, u.y, u.z, u.w };
#pragma unroll
            for (int j = 0; j < 4; ++j) {
                int e = i * 8 + j * 2;
                acc = fmaf(bf2f((unsigned short)(w[j] & 0xffffu)), arow[e], acc);
                acc = fmaf(bf2f((unsigned short)(w[j] >> 16)),    arow[e + 1], acc);
            }
        }
        out[pix * 256 + threadIdx.x] = f2bf(acc);
    }
}

// ---------------------------------------------------------------------------
// Positional path: dw3x3(x) -> exact GELU -> bf16 buffer; then dw3x3 -> += out
// ---------------------------------------------------------------------------
__global__ __launch_bounds__(256)
void pos1_kernel(const float* __restrict__ x, const float* __restrict__ w,
                 unsigned short* __restrict__ p)
{
    int pi = blockIdx.x;
    int b = pi >> 16, yx = pi & 65535, y = yx >> 8, xx0 = yx & 255;
    int c = threadIdx.x;
    float acc = 0.0f;
#pragma unroll
    for (int dy = -1; dy <= 1; ++dy) {
        int yy = y + dy; if (yy < 0 || yy > 255) continue;
#pragma unroll
        for (int dx = -1; dx <= 1; ++dx) {
            int xc = xx0 + dx; if (xc < 0 || xc > 255) continue;
            acc = fmaf(x[(((size_t)b * HW) + yy * 256 + xc) * 256 + c],
                       w[c * 9 + (dy + 1) * 3 + (dx + 1)], acc);
        }
    }
    float g = 0.5f * acc * (1.0f + erff(acc * 0.70710678118654752f));  // exact GELU
    p[(size_t)pi * 256 + c] = f2bf(g);
}

__global__ __launch_bounds__(256)
void pos2_kernel(const unsigned short* __restrict__ p, const float* __restrict__ w,
                 float* __restrict__ out)
{
    int pi = blockIdx.x;
    int b = pi >> 16, yx = pi & 65535, y = yx >> 8, xx0 = yx & 255;
    int c = threadIdx.x;
    float acc = 0.0f;
#pragma unroll
    for (int dy = -1; dy <= 1; ++dy) {
        int yy = y + dy; if (yy < 0 || yy > 255) continue;
#pragma unroll
        for (int dx = -1; dx <= 1; ++dx) {
            int xc = xx0 + dx; if (xc < 0 || xc > 255) continue;
            acc = fmaf(bf2f(p[(((size_t)b * HW) + yy * 256 + xc) * 256 + c]),
                       w[c * 9 + (dy + 1) * 3 + (dx + 1)], acc);
        }
    }
    out[(size_t)pi * 256 + c] += acc;   // out already holds the projection result
}

// ---------------------------------------------------------------------------
// Workspace layout (bytes). bf16 activation buffers: NPIX*256*2 = 64 MiB each.
// ---------------------------------------------------------------------------
static const size_t ELB    = (size_t)NPIX * CCH * 2;   // 67,108,864
static const size_t OFF_Q  = 0;
static const size_t OFF_K  = 1 * ELB;
static const size_t OFF_V  = 2 * ELB;
static const size_t OFF_M1 = 3 * ELB;                  // later reused for attn-out
static const size_t OFF_M2 = 4 * ELB;
static const size_t OFF_W  = 5 * ELB;                  // 6 * 65536 bf16 weights
static const size_t OFF_GP = OFF_W  + 6 * 65536 * 2;   // Gram partials 16*64*1024 f32
static const size_t OFF_NP = OFF_GP + (size_t)16 * 64 * 1024 * 4;  // norm partials
static const size_t OFF_AT = OFF_NP + (size_t)2 * 64 * 2 * 256 * 4; // final attn

extern "C" void kernel_launch(void* const* d_in, const int* in_sizes, int n_in,
                              void* d_out, int out_size, void* d_ws, size_t ws_size,
                              hipStream_t stream)
{
    (void)in_sizes; (void)n_in; (void)out_size; (void)ws_size;
    const float* x_in   = (const float*)d_in[0];
    const float* mask   = (const float*)d_in[1];
    const float* wq     = (const float*)d_in[2];
    const float* wk     = (const float*)d_in[3];
    const float* wv     = (const float*)d_in[4];
    const float* resc   = (const float*)d_in[5];
    const float* proj_w = (const float*)d_in[6];
    const float* proj_b = (const float*)d_in[7];
    const float* pos_w1 = (const float*)d_in[8];
    const float* pos_w2 = (const float*)d_in[9];
    const float* c1_w   = (const float*)d_in[10];
    const float* c1_b   = (const float*)d_in[11];
    const float* c2_w   = (const float*)d_in[12];
    const float* c2_b   = (const float*)d_in[13];
    const float* dw_w   = (const float*)d_in[14];
    const float* dw_b   = (const float*)d_in[15];

    char* ws = (char*)d_ws;
    unsigned short* r_q  = (unsigned short*)(ws + OFF_Q);
    unsigned short* r_k  = (unsigned short*)(ws + OFF_K);
    unsigned short* r_v  = (unsigned short*)(ws + OFF_V);
    unsigned short* r_m1 = (unsigned short*)(ws + OFF_M1);
    unsigned short* r_m2 = (unsigned short*)(ws + OFF_M2);
    unsigned short* wsW  = (unsigned short*)(ws + OFF_W);
    float*          Gp   = (float*)(ws + OFF_GP);
    float*          Np   = (float*)(ws + OFF_NP);
    float*          attn = (float*)(ws + OFF_AT);
    unsigned short* r_p  = r_q;                      // reuse q buffer for pos path
    float*          out  = (float*)d_out;

    // 1) weights -> bf16 pair-major
    wprep_kernel<<<1536, 256, 0, stream>>>(wq, wk, wv, c1_w, c2_w, proj_w, wsW);

    // 2) fused Q/K/V projections  (A = x f32, 3 B matrices resident in LDS)
    gemm_wmma<3, true, false, false, 64><<<dim3(1024, 4), 256,
        (3 * 128 * 64 + 4096) * 4, stream>>>(
        x_in, wsW, wsW + 65536, wsW + 2 * 65536,
        r_q, r_k, r_v, nullptr, nullptr, nullptr);

    // 3) mask 1x1 convs: m1 = mask @ c1^T + b ; m2 = m1 @ c2^T + b
    gemm_wmma<1, true, false, true, 128><<<dim3(1024, 2), 256,
        (128 * 128 + 4096) * 4, stream>>>(
        mask, wsW + 3 * 65536, nullptr, nullptr,
        r_m1, nullptr, nullptr, c1_b, nullptr, nullptr);
    gemm_wmma<1, false, false, true, 128><<<dim3(1024, 2), 256,
        (128 * 128 + 4096) * 4, stream>>>(
        r_m1, wsW + 4 * 65536, nullptr, nullptr,
        r_m2, nullptr, nullptr, c2_b, nullptr, nullptr);

    // 4) channel-attention scores: norms + split-K WMMA Gram + softmax
    norms_kernel<<<128, 256, 0, stream>>>(r_q, r_k, Np);
    gram_wmma<<<1024, 256, 0, stream>>>(r_q, r_k, Gp);
    softmax_kernel<<<16, 32, 0, stream>>>(Gp, Np, resc, attn);

    // 5) mask gating of v (in place), then out = attn @ v  -> r_m1 (reused)
    maskcomb_kernel<<<NPIX, 256, 0, stream>>>(r_m1, r_m2, r_v, dw_w, dw_b);
    attnapply_kernel<<<NPIX / 32, 256, 0, stream>>>(attn, r_v, r_m1);

    // 6) output projection straight to d_out (f32, +bias)
    gemm_wmma<1, false, true, true, 128><<<dim3(1024, 2), 256,
        (128 * 128 + 4096) * 4, stream>>>(
        r_m1, wsW + 5 * 65536, nullptr, nullptr,
        out, nullptr, nullptr, proj_b, nullptr, nullptr);

    // 7) positional embedding: dw3x3 -> GELU -> dw3x3, accumulated into d_out
    pos1_kernel<<<NPIX, 256, 0, stream>>>(x_in, pos_w1, r_p);
    pos2_kernel<<<NPIX, 256, 0, stream>>>(r_p, pos_w2, out);
}
